// STTransformer_50766513439028
// MI455X (gfx1250) — compile-verified, matrix operand order checked by reading, not run on hardware
//
#include <hip/hip_runtime.h>
#include <cstring>

// Model constants
#define HH   128
#define JJ   24
#define LLAY 4
#define NHH  8
#define WW   120
#define FFD  256
#define DDIM 216
#define BBAT 2
#define HDD  16
#define TOTF 121   // S + T = 60 + 61

typedef __attribute__((ext_vector_type(16))) _Float16 v16h;
typedef __attribute__((ext_vector_type(8)))  float    v8f;

// ---------------------------------------------------------------------------
// Generic strided batched GEMM:  C = postScale * act(alpha * A*B^T + bias) + R
//   A[m,k] = A[offA(z) + m*sA + k*kA]
//   B[n,k] = Bm[offB(z) + n*sB + k*kB]
//   3-level batch (z1,z2,z3) with independent strides per tensor.
// ---------------------------------------------------------------------------
struct GA {
  const float* A; const float* Bm; const float* bias; const float* R; float* C;
  long long offA, offB, offBias, offR, offC;
  int bA1,bA2,bA3, bB1,bB2,bB3, bb1,bb2,bb3, bR1,bR2,bR3, bC1,bC2,bC3;
  int sA,kA, sB,kB, sR, ldc;
  int M,N,K, nb2,nb3;
  float alpha, postScale;
  int relu, hasBias, hasRes;
};

// Load one f16 16x32 fragment from a K-contiguous, 16B-aligned row pointer.
// ISA layout: lanes 0-15 hold K {0..7,16..23}, lanes 16-31 hold K {8..15,24..31}
// (the +kb shift is already folded into the pointer by the caller).
__device__ inline v16h ldfrag16(const float* p) {
  const float4* q = (const float4*)p;
  float4 x0 = q[0];   // k+0..3
  float4 x1 = q[1];   // k+4..7
  float4 x2 = q[4];   // k+16..19
  float4 x3 = q[5];   // k+20..23
  v16h f;
  f[0]=(_Float16)x0.x;  f[1]=(_Float16)x0.y;  f[2]=(_Float16)x0.z;  f[3]=(_Float16)x0.w;
  f[4]=(_Float16)x1.x;  f[5]=(_Float16)x1.y;  f[6]=(_Float16)x1.z;  f[7]=(_Float16)x1.w;
  f[8]=(_Float16)x2.x;  f[9]=(_Float16)x2.y;  f[10]=(_Float16)x2.z; f[11]=(_Float16)x2.w;
  f[12]=(_Float16)x3.x; f[13]=(_Float16)x3.y; f[14]=(_Float16)x3.z; f[15]=(_Float16)x3.w;
  return f;
}

// Fast path: kA==kB==1, K%32==0, M%16==0, N%64==0. One wave per 16x64 C tile,
// A fragment reused across 4 WMMAs per K-chunk, b128 vector loads, no guards.
__global__ __launch_bounds__(32) void k_gemm_fast(GA g) {
  int z  = blockIdx.z;
  int z3 = z % g.nb3; int zt = z / g.nb3;
  int z2 = zt % g.nb2; int z1 = zt / g.nb2;
  long long oA = g.offA + (long long)z1*g.bA1 + (long long)z2*g.bA2 + (long long)z3*g.bA3;
  long long oB = g.offB + (long long)z1*g.bB1 + (long long)z2*g.bB2 + (long long)z3*g.bB3;
  long long oC = g.offC + (long long)z1*g.bC1 + (long long)z2*g.bC2 + (long long)z3*g.bC3;
  int m0 = blockIdx.y << 4, n0 = blockIdx.x << 6;
  int lane = threadIdx.x;
  int r  = lane & 15;
  int kb = (lane >> 4) << 3;
  const float* Ap = g.A  + oA + (long long)(m0 + r) * g.sA + kb;
  const float* Bp = g.Bm + oB + (long long)(n0 + r) * g.sB + kb;
  const long long sB16 = (long long)16 * g.sB;
  v8f c0 = {0.f,0.f,0.f,0.f,0.f,0.f,0.f,0.f};
  v8f c1 = c0, c2 = c0, c3 = c0;
#pragma unroll 2
  for (int k0 = 0; k0 < g.K; k0 += 32) {
    v16h a  = ldfrag16(Ap + k0);
    v16h b0 = ldfrag16(Bp + k0);
    v16h b1 = ldfrag16(Bp + sB16     + k0);
    v16h b2 = ldfrag16(Bp + 2*sB16   + k0);
    v16h b3 = ldfrag16(Bp + 3*sB16   + k0);
    c0 = __builtin_amdgcn_wmma_f32_16x16x32_f16(false, a, false, b0, (short)0, c0, false, false);
    c1 = __builtin_amdgcn_wmma_f32_16x16x32_f16(false, a, false, b1, (short)0, c1, false, false);
    c2 = __builtin_amdgcn_wmma_f32_16x16x32_f16(false, a, false, b2, (short)0, c2, false, false);
    c3 = __builtin_amdgcn_wmma_f32_16x16x32_f16(false, a, false, b3, (short)0, c3, false, false);
  }
  long long obias = g.offBias + (long long)z1*g.bb1 + (long long)z2*g.bb2 + (long long)z3*g.bb3;
  long long oR    = g.offR    + (long long)z1*g.bR1 + (long long)z2*g.bR2 + (long long)z3*g.bR3;
  int nbase = n0 + (lane & 15);
  int mb    = m0 + ((lane >> 4) << 3);
  auto epi = [&](v8f c, int nn) {
    float bv = g.hasBias ? g.bias[obias + nn] : 0.f;
#pragma unroll
    for (int rr = 0; rr < 8; ++rr) {
      int m = mb + rr;
      float v = g.alpha * c[rr] + bv;
      if (g.relu) v = fmaxf(v, 0.f);
      v *= g.postScale;
      if (g.hasRes) v += g.R[oR + (long long)m * g.sR + nn];
      g.C[oC + (long long)m * g.ldc + nn] = v;
    }
  };
  epi(c0, nbase); epi(c1, nbase + 16); epi(c2, nbase + 32); epi(c3, nbase + 48);
}

// Generic path: arbitrary strides / K tails. Out-of-range rows/cols are
// CLAMPED (their products land in discarded C entries), K tail is zero-
// selected after an unconditional clamped load -> branchless, batched loads.
__global__ __launch_bounds__(32) void k_gemm_gen(GA g) {
  int z  = blockIdx.z;
  int z3 = z % g.nb3; int zt = z / g.nb3;
  int z2 = zt % g.nb2; int z1 = zt / g.nb2;
  long long oA = g.offA + (long long)z1*g.bA1 + (long long)z2*g.bA2 + (long long)z3*g.bA3;
  long long oB = g.offB + (long long)z1*g.bB1 + (long long)z2*g.bB2 + (long long)z3*g.bB3;
  long long oC = g.offC + (long long)z1*g.bC1 + (long long)z2*g.bC2 + (long long)z3*g.bC3;
  int m0 = blockIdx.y << 4, n0 = blockIdx.x << 4;
  int lane = threadIdx.x;
  int r  = lane & 15;
  int kb = (lane >> 4) << 3;
  int mA = m0 + r; if (mA > g.M - 1) mA = g.M - 1;   // clamp, not guard
  int nB = n0 + r; if (nB > g.N - 1) nB = g.N - 1;
  const float* Ap = g.A  + oA + (long long)mA * g.sA;
  const float* Bp = g.Bm + oB + (long long)nB * g.sB;
  int Km1 = g.K - 1;
  v8f c = {0.f,0.f,0.f,0.f,0.f,0.f,0.f,0.f};
  for (int k0 = 0; k0 < g.K; k0 += 32) {
    v16h a, b;
#pragma unroll
    for (int i = 0; i < 8; ++i) {
      int k1 = k0 + kb + i, k2 = k1 + 16;
      int k1c = k1 > Km1 ? Km1 : k1;
      int k2c = k2 > Km1 ? Km1 : k2;
      float a1 = Ap[(long long)k1c * g.kA];
      float a2 = Ap[(long long)k2c * g.kA];
      float b1 = Bp[(long long)k1c * g.kB];
      float b2 = Bp[(long long)k2c * g.kB];
      a[i]   = (k1 <= Km1) ? (_Float16)a1 : (_Float16)0.f;
      a[i+8] = (k2 <= Km1) ? (_Float16)a2 : (_Float16)0.f;
      b[i]   = (k1 <= Km1) ? (_Float16)b1 : (_Float16)0.f;
      b[i+8] = (k2 <= Km1) ? (_Float16)b2 : (_Float16)0.f;
    }
    c = __builtin_amdgcn_wmma_f32_16x16x32_f16(false, a, false, b, (short)0, c, false, false);
  }
  int nn = n0 + (lane & 15);
  int mb = m0 + ((lane >> 4) << 3);
  if (nn < g.N) {
    float bv = 0.f;
    if (g.hasBias)
      bv = g.bias[g.offBias + (long long)z1*g.bb1 + (long long)z2*g.bb2 + (long long)z3*g.bb3 + nn];
    long long oR = g.offR + (long long)z1*g.bR1 + (long long)z2*g.bR2 + (long long)z3*g.bR3;
#pragma unroll
    for (int rr = 0; rr < 8; ++rr) {
      int m = mb + rr;
      if (m < g.M) {
        float v = g.alpha * c[rr] + bv;
        if (g.relu) v = fmaxf(v, 0.f);
        v *= g.postScale;
        if (g.hasRes) v += g.R[oR + (long long)m * g.sR + nn];
        g.C[oC + (long long)m * g.ldc + nn] = v;
      }
    }
  }
}

// ---------------------------------------------------------------------------
// Softmax over rows (optionally causal: row q keeps keys 0..q, zeros the rest
// so the following dense probs@V WMMA GEMM is exact).
// ---------------------------------------------------------------------------
__global__ void k_softmax(float* S, int nrows, int ncols, int rstride, int qmod, int causal) {
  int rid = blockIdx.x * blockDim.x + threadIdx.x;
  if (rid >= nrows) return;
  int len = causal ? (rid % qmod) + 1 : ncols;
  float* row = S + (long long)rid * rstride;
  float mx = -3.0e38f;
  for (int i = 0; i < len; ++i) mx = fmaxf(mx, row[i]);
  float s = 0.f;
  for (int i = 0; i < len; ++i) { float e = __expf(row[i] - mx); row[i] = e; s += e; }
  float inv = 1.f / s;
  for (int i = 0; i < len; ++i) row[i] *= inv;
  for (int i = len; i < ncols; ++i) row[i] = 0.f;
}

// One wave32 per 128-wide token; shfl_xor tree reduction (CDNA5 wave32).
__global__ __launch_bounds__(32) void k_layernorm(float* X, const float* gg, const float* bb) {
  long long t = blockIdx.x;
  int lane = threadIdx.x;
  float v0 = X[t*128 + lane], v1 = X[t*128 + lane + 32];
  float v2 = X[t*128 + lane + 64], v3 = X[t*128 + lane + 96];
  float s = v0 + v1 + v2 + v3;
  for (int m = 16; m > 0; m >>= 1) s += __shfl_xor(s, m, 32);
  float mean = s * (1.f/128.f);
  float d0=v0-mean, d1=v1-mean, d2=v2-mean, d3=v3-mean;
  float var = d0*d0 + d1*d1 + d2*d2 + d3*d3;
  for (int m = 16; m > 0; m >>= 1) var += __shfl_xor(var, m, 32);
  float rstd = rsqrtf(var*(1.f/128.f) + 1e-5f);
  X[t*128 + lane]      = d0*rstd*gg[lane]      + bb[lane];
  X[t*128 + lane + 32] = d1*rstd*gg[lane + 32] + bb[lane + 32];
  X[t*128 + lane + 64] = d2*rstd*gg[lane + 64] + bb[lane + 64];
  X[t*128 + lane + 96] = d3*rstd*gg[lane + 96] + bb[lane + 96];
}

// concat[tt,b,:] = tt<60 ? src[b,tt,:] : tgt[b,tt-60,:]
__global__ void k_concat(const float* src, const float* tgt, float* cc) {
  int i = blockIdx.x * 256 + threadIdx.x;
  if (i >= TOTF*BBAT*DDIM) return;
  int dd = i % DDIM; int m = i / DDIM; int b = m & 1; int tt = m >> 1;
  cc[i] = (tt < 60) ? src[(b*60 + tt)*DDIM + dd] : tgt[(b*61 + (tt - 60))*DDIM + dd];
}

// sinusoidal positional encoding added in-place on x0 (m = tt*B + b, e = j*128+h)
__global__ void k_peadd(float* x0) {
  long long i = (long long)blockIdx.x * 256 + threadIdx.x;
  if (i >= (long long)TOTF*BBAT*JJ*HH) return;
  int e  = (int)(i % (JJ*HH));
  int m  = (int)(i / (JJ*HH));
  int tt = m >> 1;
  float freq = __expf(((e >> 1) * 2) * (-9.2103403719761836f / (float)(JJ*HH)));
  float a = (float)tt * freq;
  x0[i] += (e & 1) ? __cosf(a) : __sinf(a);
}

// tout[t,b,j,:] += sp_out[b,j,:]
__global__ void k_addsp(float* tout, const float* sp, int t) {
  int i = blockIdx.x * 256 + threadIdx.x;
  if (i >= BBAT*JJ*HH) return;
  int hh = i & 127; int m = i >> 7; int b = m / JJ; int j = m % JJ;
  tout[((((long long)t*BBAT + b)*JJ) + j)*HH + hh] += sp[i];
}

// out[b,t,:] = x[t] @ Wout^T + bout + concat[t]   (N=9, too small for WMMA)
__global__ void k_final(const float* x, const float* Wout, const float* bout,
                        const float* cc, float* out, int t) {
  int i = blockIdx.x * 256 + threadIdx.x;
  if (i >= BBAT*JJ*9) return;
  int o = i % 9; int m = i / 9; int b = m / JJ; int j = m % JJ;
  const float* xr = x + ((((long long)t*BBAT + b)*JJ) + j)*HH;
  const float* wr = Wout + o*HH;
  float s = bout[o];
  for (int h = 0; h < HH; ++h) s += xr[h] * wr[h];
  s += cc[((long long)t*BBAT + b)*DDIM + j*9 + o];
  out[((long long)b*WW + t)*DDIM + j*9 + o] = s;
}

// ---------------------------------------------------------------------------
static GA ga0() {
  GA g; std::memset(&g, 0, sizeof g);
  g.kA = 1; g.kB = 1; g.alpha = 1.f; g.postScale = 1.f; g.nb2 = 1; g.nb3 = 1;
  return g;
}
static void glaunch(hipStream_t st, const GA& g, int nb1) {
  bool fast = (g.kA == 1) && (g.kB == 1) && ((g.K & 31) == 0) &&
              ((g.M & 15) == 0) && ((g.N & 63) == 0);
  if (fast) {
    dim3 grid((unsigned)(g.N >> 6), (unsigned)(g.M >> 4),
              (unsigned)(nb1 * g.nb2 * g.nb3));
    k_gemm_fast<<<grid, dim3(32), 0, st>>>(g);
  } else {
    dim3 grid((unsigned)((g.N + 15) / 16), (unsigned)((g.M + 15) / 16),
              (unsigned)(nb1 * g.nb2 * g.nb3));
    k_gemm_gen<<<grid, dim3(32), 0, st>>>(g);
  }
}

extern "C" void kernel_launch(void* const* d_in, const int* in_sizes, int n_in,
                              void* d_out, int out_size, void* d_ws, size_t ws_size,
                              hipStream_t stream) {
  (void)in_sizes; (void)n_in; (void)out_size; (void)ws_size;
  const float* src    = (const float*)d_in[0];
  const float* tgt    = (const float*)d_in[1];
  const float* Wemb   = (const float*)d_in[2];
  const float* bemb   = (const float*)d_in[3];
  const float* Wqkv_s = (const float*)d_in[4];
  const float* bqkv_s = (const float*)d_in[5];
  const float* Wo_s   = (const float*)d_in[6];
  const float* bo_s   = (const float*)d_in[7];
  const float* sn_g   = (const float*)d_in[8];
  const float* sn_b   = (const float*)d_in[9];
  const float* Wqkv_t = (const float*)d_in[10];
  const float* bqkv_t = (const float*)d_in[11];
  const float* Wo_t   = (const float*)d_in[12];
  const float* bo_t   = (const float*)d_in[13];
  const float* tn_g   = (const float*)d_in[14];
  const float* tn_b   = (const float*)d_in[15];
  const float* W1     = (const float*)d_in[16];
  const float* b1     = (const float*)d_in[17];
  const float* W2     = (const float*)d_in[18];
  const float* b2     = (const float*)d_in[19];
  const float* Wout   = (const float*)d_in[20];
  const float* bout   = (const float*)d_in[21];
  float* out = (float*)d_out;

  // workspace carve-up (floats), ~52 MB total
  float* w = (float*)d_ws;
  float* ccat  = w; w += TOTF*BBAT*DDIM;               // 52272
  float* x0    = w; w += (long long)TOTF*BBAT*JJ*HH;   // 743424
  float* xA    = w; w += (long long)WW*BBAT*JJ*HH;     // 737280
  float* xB    = w; w += (long long)WW*BBAT*JJ*HH;
  float* tout  = w; w += (long long)WW*BBAT*JJ*HH;
  float* ffn   = w; w += (long long)WW*BBAT*JJ*FFD;    // 1474560
  float* qkv_s = w; w += BBAT*JJ*3*HH;                 // 18432
  float* sc_s  = w; w += BBAT*NHH*JJ*JJ;               // 9216
  float* o_s   = w; w += BBAT*JJ*HH;                   // 6144
  float* sp    = w; w += BBAT*JJ*HH;                   // 6144
  float* qkv_t = w; w += (long long)JJ*WW*BBAT*3*HH;   // 2211840
  float* sc_t  = w; w += (long long)JJ*BBAT*NHH*WW*WW; // 5529600
  float* ao_t  = w; w += (long long)JJ*WW*BBAT*HH;     // 737280

  // 1) concat src/tgt -> (121,B,216)
  k_concat<<<(TOTF*BBAT*DDIM + 255)/256, 256, 0, stream>>>(src, tgt, ccat);

  // 2) embedding GEMM: (242,216) @ (3072,216)^T + bemb -> x0 (121,B,24,128)
  {
    GA g = ga0();
    g.A = ccat; g.sA = DDIM;
    g.Bm = Wemb; g.sB = DDIM;
    g.bias = bemb; g.hasBias = 1;
    g.C = x0; g.ldc = JJ*HH;
    g.M = TOTF*BBAT; g.N = JJ*HH; g.K = DDIM;
    glaunch(stream, g, 1);
  }
  // 3) positional encoding
  k_peadd<<<(int)(((long long)TOTF*BBAT*JJ*HH + 255)/256), 256, 0, stream>>>(x0);

  // 4) scan over t (iterations independent; layer0 temporal attn is t-invariant
  //    and could additionally be hoisted for a further ~25% FLOP cut)
  for (int t = 0; t < WW; ++t) {
    const float* xin = x0;
    float* xout = xA;
    for (int l = 0; l < LLAY; ++l) {
      xout = (l & 1) ? xB : xA;

      // ---- spatial attention (tokens m = b*24+j on frame t) ----
      { // qkv_s = x[t] @ Wqkv_s[l]^T + b                          [fast]
        GA g = ga0();
        g.A = xin; g.offA = (long long)t*BBAT*JJ*HH; g.sA = HH;
        g.Bm = Wqkv_s; g.offB = (long long)l*3*HH*HH; g.sB = HH;
        g.bias = bqkv_s; g.offBias = (long long)l*3*HH; g.hasBias = 1;
        g.C = qkv_s; g.ldc = 3*HH; g.M = BBAT*JJ; g.N = 3*HH; g.K = HH;
        glaunch(stream, g, 1);
      }
      { // scores per (b,h): Q @ K^T * 0.25, M=N=24, K=16          [gen]
        GA g = ga0();
        g.A = qkv_s; g.sA = 3*HH; g.bA2 = JJ*3*HH; g.bA3 = HDD;
        g.Bm = qkv_s; g.offB = HH; g.sB = 3*HH; g.bB2 = JJ*3*HH; g.bB3 = HDD;
        g.C = sc_s; g.ldc = JJ; g.bC2 = NHH*JJ*JJ; g.bC3 = JJ*JJ;
        g.M = JJ; g.N = JJ; g.K = HDD; g.alpha = 0.25f;
        g.nb2 = BBAT; g.nb3 = NHH;
        glaunch(stream, g, 1);
      }
      k_softmax<<<(BBAT*NHH*JJ + 255)/256, 256, 0, stream>>>(sc_s, BBAT*NHH*JJ, JJ, JJ, 1, 0);
      { // o_s per (b,h): probs @ V (V column-accessed via kB)     [gen]
        GA g = ga0();
        g.A = sc_s; g.sA = JJ; g.bA2 = NHH*JJ*JJ; g.bA3 = JJ*JJ;
        g.Bm = qkv_s; g.offB = 2*HH; g.sB = 1; g.kB = 3*HH; g.bB2 = JJ*3*HH; g.bB3 = HDD;
        g.C = o_s; g.ldc = HH; g.bC2 = JJ*HH; g.bC3 = HDD;
        g.M = JJ; g.N = HDD; g.K = JJ; g.nb2 = BBAT; g.nb3 = NHH;
        glaunch(stream, g, 1);
      }
      { // sp = o_s @ Wo_s^T + bo + x[t], then LN                  [fast]
        GA g = ga0();
        g.A = o_s; g.sA = HH;
        g.Bm = Wo_s; g.offB = (long long)l*HH*HH; g.sB = HH;
        g.bias = bo_s; g.offBias = (long long)l*HH; g.hasBias = 1;
        g.R = xin; g.offR = (long long)t*BBAT*JJ*HH; g.sR = HH; g.hasRes = 1;
        g.C = sp; g.ldc = HH; g.M = BBAT*JJ; g.N = HH; g.K = HH;
        glaunch(stream, g, 1);
      }
      k_layernorm<<<BBAT*JJ, 32, 0, stream>>>(sp, sn_g + l*HH, sn_b + l*HH);

      // ---- temporal attention (per joint j, tokens m = w*B+b) ----
      { // qkv_t[j] = temp[j] @ Wqkv_t[l,j]^T + b (batched j)      [fast]
        GA g = ga0();
        g.A = xin; g.sA = JJ*HH; g.bA1 = HH;
        g.Bm = Wqkv_t; g.offB = (long long)l*JJ*3*HH*HH; g.bB1 = 3*HH*HH; g.sB = HH;
        g.bias = bqkv_t; g.offBias = (long long)l*JJ*3*HH; g.bb1 = 3*HH; g.hasBias = 1;
        g.C = qkv_t; g.ldc = 3*HH; g.bC1 = WW*BBAT*3*HH;
        g.M = WW*BBAT; g.N = 3*HH; g.K = HH;
        glaunch(stream, g, JJ);
      }
      { // scores per (j,b,h): 120x120, K=16, *0.25                [gen]
        GA g = ga0();
        g.A = qkv_t; g.sA = BBAT*3*HH; g.bA1 = WW*BBAT*3*HH; g.bA2 = 3*HH; g.bA3 = HDD;
        g.Bm = qkv_t; g.offB = HH; g.sB = BBAT*3*HH; g.bB1 = WW*BBAT*3*HH; g.bB2 = 3*HH; g.bB3 = HDD;
        g.C = sc_t; g.ldc = WW; g.bC1 = BBAT*NHH*WW*WW; g.bC2 = NHH*WW*WW; g.bC3 = WW*WW;
        g.M = WW; g.N = WW; g.K = HDD; g.alpha = 0.25f;
        g.nb2 = BBAT; g.nb3 = NHH;
        glaunch(stream, g, JJ);
      }
      // causal softmax (masked keys -> 0 so dense AV GEMM stays exact)
      k_softmax<<<(JJ*BBAT*NHH*WW + 255)/256, 256, 0, stream>>>(
          sc_t, JJ*BBAT*NHH*WW, WW, WW, WW, 1);
      { // ao_t per (j,b,h): probs(120x120) @ V(120x16)            [gen]
        GA g = ga0();
        g.A = sc_t; g.sA = WW; g.bA1 = BBAT*NHH*WW*WW; g.bA2 = NHH*WW*WW; g.bA3 = WW*WW;
        g.Bm = qkv_t; g.offB = 2*HH; g.sB = 1; g.kB = BBAT*3*HH;
        g.bB1 = WW*BBAT*3*HH; g.bB2 = 3*HH; g.bB3 = HDD;
        g.C = ao_t; g.ldc = BBAT*HH; g.bC1 = WW*BBAT*HH; g.bC2 = HH; g.bC3 = HDD;
        g.M = WW; g.N = HDD; g.K = WW; g.nb2 = BBAT; g.nb3 = NHH;
        glaunch(stream, g, JJ);
      }
      { // tout = ao_t @ Wo_t[l,j]^T + bo + temp                   [fast]
        GA g = ga0();
        g.A = ao_t; g.sA = HH; g.bA1 = WW*BBAT*HH;
        g.Bm = Wo_t; g.offB = (long long)l*JJ*HH*HH; g.bB1 = HH*HH; g.sB = HH;
        g.bias = bo_t; g.offBias = (long long)l*JJ*HH; g.bb1 = HH; g.hasBias = 1;
        g.R = xin; g.sR = JJ*HH; g.bR1 = HH; g.hasRes = 1;
        g.C = tout; g.ldc = JJ*HH; g.bC1 = HH;
        g.M = WW*BBAT; g.N = HH; g.K = HH;
        glaunch(stream, g, JJ);
      }
      k_layernorm<<<WW*BBAT*JJ, 32, 0, stream>>>(tout, tn_g + l*HH, tn_b + l*HH);
      k_addsp<<<(BBAT*JJ*HH + 255)/256, 256, 0, stream>>>(tout, sp, t);

      // ---- FFN: x_next = (2*relu(tout@W1^T+b1)) @ W2^T + b2 ----  [fast x2]
      {
        GA g = ga0();
        g.A = tout; g.sA = HH;
        g.Bm = W1; g.offB = (long long)l*FFD*HH; g.sB = HH;
        g.bias = b1; g.offBias = (long long)l*FFD; g.hasBias = 1;
        g.relu = 1; g.postScale = 2.f;
        g.C = ffn; g.ldc = FFD; g.M = WW*BBAT*JJ; g.N = FFD; g.K = HH;
        glaunch(stream, g, 1);
      }
      {
        GA g = ga0();
        g.A = ffn; g.sA = FFD;
        g.Bm = W2; g.offB = (long long)l*HH*FFD; g.sB = FFD;
        g.bias = b2; g.offBias = (long long)l*HH; g.hasBias = 1;
        g.C = xout; g.ldc = HH; g.M = WW*BBAT*JJ; g.N = HH; g.K = FFD;
        glaunch(stream, g, 1);
      }
      xin = xout;
    }
    // final projection + input residual for step t
    k_final<<<(BBAT*JJ*9 + 255)/256, 256, 0, stream>>>(xin, Wout, bout, ccat, out, t);
  }
}
